// Decoder_62852551409990
// MI455X (gfx1250) — matrix operand order repaired and checked by reading, not compiled, not run on hardware
//
#include <hip/hip_runtime.h>
#include <stdint.h>

// ---------------- types ----------------
typedef __bf16        v16bf __attribute__((ext_vector_type(16)));
typedef __bf16        v4bf  __attribute__((ext_vector_type(4)));
typedef __bf16        v2bf  __attribute__((ext_vector_type(2)));
typedef float         v8f   __attribute__((ext_vector_type(8)));
typedef float         f32x2 __attribute__((ext_vector_type(2)));
typedef float         f32x4 __attribute__((ext_vector_type(4)));
typedef unsigned int  u32x8 __attribute__((ext_vector_type(8)));
typedef unsigned int  u32x4 __attribute__((ext_vector_type(4)));
typedef int           i32x4 __attribute__((ext_vector_type(4)));
typedef int           i32x8 __attribute__((ext_vector_type(8)));

#define VOCAB 50257
#define EMB   256
#define HID   256
#define BATCH 128
#define SEQ   2048

// LDS tile: 16 rows x 256 cols of bf16, row stride padded to 268 halfs
// (268/2 = 134 dwords, 134 % 64 = 6 -> conflict-free fragment reads,
//  row stride 536 B is 8B-aligned for b64 stores)
#define TSTRIDE 268

// workspace layout (float offsets)
#define OFF_U      0           // 128*256
#define OFF_E      32768       // 128*2048 attention logits [b][s]
#define OFF_CTXP   294912      // 8*128*256 ctx partials
#define OFF_RNN    557056      // 128*512
#define OFF_GATES  622592      // 128*1024
#define OFF_HNEW   753664      // 128*256
#define OFF_W1P    786432      // 131072 B : W1 B-fragments
#define OFF_APACK  819200      // 65536 B  : h_new A-fragments

#if defined(__has_builtin)
#  if __has_builtin(__builtin_amdgcn_tensor_load_to_lds) && \
      __has_builtin(__builtin_amdgcn_s_wait_tensorcnt)
#    define HAVE_TDM 1
#  endif
#endif
#ifndef HAVE_TDM
#  define HAVE_TDM 0
#endif

// ---------------- helpers ----------------
// f32 -> bf16 (RNE) through vector fptrunc: selects v_cvt_pk_bf16_f32 on gfx1250
__device__ __forceinline__ unsigned pack2(float a, float b) {
  f32x2 v = {a, b};
  return __builtin_bit_cast(unsigned, __builtin_convertvector(v, v2bf));
}

__device__ __forceinline__ v8f wmma_bf16(v16bf a, v16bf b, v8f c) {
  // D = A(16x32 bf16) * B(32x16 bf16) + C(16x16 f32)
  return __builtin_amdgcn_wmma_f32_16x16x32_bf16(false, a, false, b,
                                                 (short)0, c, false, false);
}

// stage 16x256 f32 tile (rows at src + row*srcStride) -> bf16 LDS tile
__device__ __forceinline__ void stage_tile(unsigned short* buf, const float* src,
                                           long srcStride, int lane) {
#pragma unroll
  for (int it = 0; it < 32; ++it) {
    int idx = (it * 32 + lane) * 4;
    int row = idx >> 8;
    int col = idx & 255;
    f32x4 v = *reinterpret_cast<const f32x4*>(src + (long)row * srcStride + col);
    v4bf  r = __builtin_convertvector(v, v4bf);
    *reinterpret_cast<v4bf*>(&buf[row * TSTRIDE + col]) = r;
  }
}

// A fragment (16x32 bf16) from LDS tile: VGPR i holds K = 2(i&3)+(i>>2)*16 (+8 hi lanes)
__device__ __forceinline__ v16bf afrag_lds(const unsigned short* buf, int kt, int lane) {
  int m   = lane & 15;
  int khi = (lane & 16) ? 8 : 0;
  u32x8 d;
#pragma unroll
  for (int i = 0; i < 8; ++i) {
    int k = kt * 32 + (i & 3) * 2 + (i >> 2) * 16 + khi;
    d[i] = *reinterpret_cast<const unsigned int*>(&buf[m * TSTRIDE + k]);
  }
  return __builtin_bit_cast(v16bf, d);
}

// B fragment (32x16 bf16) from LDS tile holding rows n (B[k][n] = W[n][k])
__device__ __forceinline__ v16bf bfrag_lds(const unsigned short* buf, int kt, int lane) {
  int n  = lane & 15;
  int khi = (lane & 16) ? 8 : 0;
  u32x8 d;
#pragma unroll
  for (int i = 0; i < 8; ++i)
    d[i] = *reinterpret_cast<const unsigned int*>(&buf[n * TSTRIDE + kt * 32 + (i & 3) * 2 + (i >> 2) * 16 + khi]);
  return __builtin_bit_cast(v16bf, d);
}

// A fragment straight from global f32 (small, L2-hot matrices)
__device__ __forceinline__ v16bf afrag_glb(const float* A, long AR, int rbase,
                                           int c0, int kt, int lane) {
  int m   = rbase + (lane & 15);
  int khi = (lane & 16) ? 8 : 0;
  u32x8 d;
#pragma unroll
  for (int i = 0; i < 8; ++i) {
    int k = kt * 32 + (i & 3) * 2 + (i >> 2) * 16 + khi;
    f32x2 v = *reinterpret_cast<const f32x2*>(A + (long)m * AR + c0 + k);
    d[i] = pack2(v[0], v[1]);
  }
  return __builtin_bit_cast(v16bf, d);
}

// B fragment straight from global f32 weight rows (B[k][n] = W[n][k])
__device__ __forceinline__ v16bf bfrag_glb(const float* W, long WR, int nbase,
                                           int c0, int kt, int lane) {
  int n  = nbase + (lane & 15);
  int khi = (lane & 16) ? 8 : 0;
  u32x8 d;
#pragma unroll
  for (int i = 0; i < 8; ++i) {
    f32x2 v = *reinterpret_cast<const f32x2*>(W + (long)n * WR + c0 + kt * 32 + (i & 3) * 2 + (i >> 2) * 16 + khi);
    d[i] = pack2(v[0], v[1]);
  }
  return __builtin_bit_cast(v16bf, d);
}

#if HAVE_TDM
// 1D TDM copy: nElem contiguous dwords global -> LDS (D# per CDNA5 ISA ch.8)
__device__ __forceinline__ void tdm_load_row(void* ldsDst, const void* gsrc,
                                             unsigned nElem) {
  unsigned long long ga = (unsigned long long)gsrc;
  unsigned lds = (unsigned)(unsigned long long)ldsDst;  // low 32 bits = LDS offset
  u32x4 g0 = { 1u,                                       // count=1, user mode
               lds,                                      // lds_addr
               (unsigned)ga,                             // global_addr[31:0]
               (unsigned)((ga >> 32) & 0x1FFFFFFu) | (2u << 30) }; // [56:32] | type=2
  i32x8 g1 = { (int)(2u << 16),          // workgroup_mask=0, data_size=4B
               (int)(nElem << 16),       // tensor_dim0[15:0]
               (int)(1u << 16),          // tensor_dim0[31:16]=0 | tensor_dim1=1
               (int)(nElem << 16),       // tensor_dim1 hi=0 | tile_dim0=nElem
               1,                        // tile_dim1=1, tile_dim2=0
               (int)nElem,               // tensor_dim0_stride[31:0]
               0, 0 };
  i32x4 gz = { 0, 0, 0, 0 };
#  if __clang_major__ >= 23
  i32x8 gz8 = { 0, 0, 0, 0, 0, 0, 0, 0 };
  __builtin_amdgcn_tensor_load_to_lds(g0, g1, gz, gz, gz8, 0);
#  else
  __builtin_amdgcn_tensor_load_to_lds(g0, g1, gz, gz, 0);
#  endif
}
#endif

// ---------------- kernel 0b: pack W1 into B-fragment layout ----------------
__global__ void k_packW1(const float* __restrict__ W1, u32x8* __restrict__ w1p) {
  int t    = blockIdx.x * blockDim.x + threadIdx.x;   // 4096 threads
  int lane = t & 31;
  int nt   = (t >> 5) & 15;
  int kt   = t >> 9;
  int n  = nt * 16 + (lane & 15);
  int khi = (lane & 16) ? 8 : 0;
  u32x8 d;
#pragma unroll
  for (int i = 0; i < 8; ++i) {
    f32x2 v = *reinterpret_cast<const f32x2*>(W1 + (long)n * HID + kt * 32 + (i & 3) * 2 + (i >> 2) * 16 + khi);
    d[i] = pack2(v[0], v[1]);
  }
  w1p[t] = d;
}

// ---------------- kernel 0a: u = h0 @ W2^T + W2_b + W1_b ----------------
__global__ void __launch_bounds__(128) k_u(const float* __restrict__ h0,
                                           const float* __restrict__ W2,
                                           const float* __restrict__ W2b,
                                           const float* __restrict__ W1b,
                                           float* __restrict__ uout) {
  int lane = threadIdx.x & 31;
  int w    = blockIdx.x * 4 + (threadIdx.x >> 5);   // 128 waves
  int mt = w >> 4, nt = w & 15;
  v8f acc = {};
#pragma unroll
  for (int kt = 0; kt < 8; ++kt) {
    v16bf a = afrag_glb(h0, HID, mt * 16, 0, kt, lane);
    v16bf b = bfrag_glb(W2, HID, nt * 16, 0, kt, lane);
    acc = wmma_bf16(a, b, acc);
  }
  int n = nt * 16 + (lane & 15);
  float bias = W2b[n] + W1b[n];
  int bhalf = (lane & 16) ? 8 : 0;
#pragma unroll
  for (int j = 0; j < 8; ++j)
    uout[(mt * 16 + j + bhalf) * HID + n] = acc[j] + bias;
}

// ------- kernel 1: fused e[b,s] = V . tanh(enc@W1^T + u) + Vb (big GEMM) -------
__global__ void __launch_bounds__(128) k_attn_score(const float* __restrict__ enc,
                                                    const float* __restrict__ uws,
                                                    const u32x8* __restrict__ w1p,
                                                    const float* __restrict__ Vw,
                                                    const float* __restrict__ Vb,
                                                    float* __restrict__ eBuf) {
  __shared__ unsigned short tile[4][16 * TSTRIDE];
  int lane = threadIdx.x & 31;
  int widx = threadIdx.x >> 5;
  int s    = blockIdx.x >> 1;                 // row block = 64 consecutive rows, one s
  int bbase = (blockIdx.x & 1) * 64 + widx * 16;
  long rw   = (long)blockIdx.x * 64 + widx * 16;   // rows r = s*128 + b, contiguous
  unsigned short* buf = tile[widx];
  stage_tile(buf, enc + rw * 256, 256, lane); // 16 contiguous rows = contiguous 16KB

  float esum[8];
#pragma unroll
  for (int j = 0; j < 8; ++j) esum[j] = 0.f;
  int bhalf = (lane & 16) ? 8 : 0;

  for (int g = 0; g < 2; ++g) {               // two groups of 8 N-tiles
    v8f acc[8];
    v8f zero = {};
#pragma unroll
    for (int q = 0; q < 8; ++q) acc[q] = zero;
#pragma unroll
    for (int kt = 0; kt < 8; ++kt) {
      v16bf a = afrag_lds(buf, kt, lane);
#pragma unroll
      for (int q = 0; q < 8; ++q) {
        int nt = g * 8 + q;
        v16bf b = __builtin_bit_cast(v16bf, w1p[(kt * 16 + nt) * 32 + lane]);
        acc[q] = wmma_bf16(a, b, acc[q]);
      }
    }
#pragma unroll
    for (int q = 0; q < 8; ++q) {
      int nt = g * 8 + q;
      int n  = nt * 16 + (lane & 15);
      float vv = Vw[n];
#pragma unroll
      for (int j = 0; j < 8; ++j) {
        int bb  = bbase + j + bhalf;
        float t = tanhf(acc[q][j] + uws[bb * HID + n]);
        esum[j] = fmaf(vv, t, esum[j]);
      }
    }
  }
  float vb = Vb[0];
#pragma unroll
  for (int j = 0; j < 8; ++j) {
    float v = esum[j];
    v += __shfl_xor(v, 8, 32);
    v += __shfl_xor(v, 4, 32);
    v += __shfl_xor(v, 2, 32);
    v += __shfl_xor(v, 1, 32);
    if ((lane & 15) == 0) {
      int bb = bbase + j + bhalf;
      eBuf[bb * SEQ + s] = v + vb;
    }
  }
}

// ---------------- kernel 2: softmax over S per batch row ----------------
__global__ void __launch_bounds__(256) k_softmax(const float* __restrict__ eBuf,
                                                 float* __restrict__ attw) {
  int b = blockIdx.x, tid = threadIdx.x;
  __shared__ float red[8];
  float v[8];
#pragma unroll
  for (int i = 0; i < 8; ++i) v[i] = eBuf[b * SEQ + tid + 256 * i];
  float m = v[0];
#pragma unroll
  for (int i = 1; i < 8; ++i) m = fmaxf(m, v[i]);
#pragma unroll
  for (int off = 16; off; off >>= 1) m = fmaxf(m, __shfl_xor(m, off, 32));
  if ((tid & 31) == 0) red[tid >> 5] = m;
  __syncthreads();
  m = red[0];
#pragma unroll
  for (int i = 1; i < 8; ++i) m = fmaxf(m, red[i]);
  float sum = 0.f;
#pragma unroll
  for (int i = 0; i < 8; ++i) { v[i] = __expf(v[i] - m); sum += v[i]; }
#pragma unroll
  for (int off = 16; off; off >>= 1) sum += __shfl_xor(sum, off, 32);
  __syncthreads();
  if ((tid & 31) == 0) red[tid >> 5] = sum;
  __syncthreads();
  sum = 0.f;
#pragma unroll
  for (int i = 0; i < 8; ++i) sum += red[i];
  float inv = 1.f / sum;
#pragma unroll
  for (int i = 0; i < 8; ++i) attw[b * SEQ + tid + 256 * i] = v[i] * inv;
}

// ------- kernel 3: ctx partials (deterministic 2-stage, TDM attw staging) -------
__global__ void __launch_bounds__(256) k_ctx_partial(const float* __restrict__ enc,
                                                     const float* __restrict__ attw,
                                                     float* __restrict__ partial) {
  int b = blockIdx.x >> 3, ch = blockIdx.x & 7;
  int h = threadIdx.x;
  __shared__ float p[256];
#if HAVE_TDM
  if (threadIdx.x < 32) {                      // wave 0 issues the DMA
    tdm_load_row(p, attw + b * SEQ + ch * 256, 256);
    __builtin_amdgcn_s_wait_tensorcnt(0);
  }
  __syncthreads();
#else
  p[h] = attw[b * SEQ + ch * 256 + h];
  __syncthreads();
#endif
  float acc = 0.f;
  const float* base = enc + (long)(ch * 256) * (BATCH * HID) + b * HID + h;
#pragma unroll 4
  for (int si = 0; si < 256; ++si) {
    if (si + 4 < 256)
      __builtin_prefetch(base + (long)(si + 4) * (BATCH * HID), 0, 1);
    acc = fmaf(p[si], base[(long)si * (BATCH * HID)], acc);
  }
  partial[(ch * BATCH + b) * HID + h] = acc;
}

// ---------------- kernel 4: rnn_in = concat(ctx, emb[x]) ----------------
__global__ void k_rnn_in(const float* __restrict__ partial, const float* __restrict__ emb,
                         const int* __restrict__ x, float* __restrict__ rnn) {
  int b = blockIdx.x, h = threadIdx.x;
  float acc = 0.f;
#pragma unroll
  for (int ch = 0; ch < 8; ++ch) acc += partial[(ch * BATCH + b) * HID + h];
  rnn[b * 512 + h]       = acc;
  rnn[b * 512 + 256 + h] = emb[(long)x[b] * EMB + h];
}

// ---------------- kernel 5: gates = rnn@W_ih^T + h0@W_hh^T + biases ----------------
__global__ void __launch_bounds__(128) k_gates(const float* __restrict__ rnn,
                                               const float* __restrict__ h0,
                                               const float* __restrict__ Wih,
                                               const float* __restrict__ Whh,
                                               const float* __restrict__ bih,
                                               const float* __restrict__ bhh,
                                               float* __restrict__ gates) {
  __shared__ unsigned short tile[4][16 * TSTRIDE];
  int lane = threadIdx.x & 31;
  int widx = threadIdx.x >> 5;
  int nt   = blockIdx.x * 4 + widx;           // 0..63 (N = 1024)
  unsigned short* buf = tile[widx];
  v8f acc[8];
  v8f zero = {};
#pragma unroll
  for (int q = 0; q < 8; ++q) acc[q] = zero;

  // part 1: K in [0,256): A = rnn[:,0:256], B = W_ih[:,0:256]
  stage_tile(buf, Wih + (long)(nt * 16) * 512, 512, lane);
#pragma unroll
  for (int kt = 0; kt < 8; ++kt) {
    v16bf bb = bfrag_lds(buf, kt, lane);
#pragma unroll
    for (int mt = 0; mt < 8; ++mt)
      acc[mt] = wmma_bf16(afrag_glb(rnn, 512, mt * 16, 0, kt, lane), bb, acc[mt]);
  }
  // part 2: K in [256,512)
  stage_tile(buf, Wih + (long)(nt * 16) * 512 + 256, 512, lane);
#pragma unroll
  for (int kt = 0; kt < 8; ++kt) {
    v16bf bb = bfrag_lds(buf, kt, lane);
#pragma unroll
    for (int mt = 0; mt < 8; ++mt)
      acc[mt] = wmma_bf16(afrag_glb(rnn, 512, mt * 16, 256, kt, lane), bb, acc[mt]);
  }
  // part 3: + h0 @ W_hh^T
  stage_tile(buf, Whh + (long)(nt * 16) * 256, 256, lane);
#pragma unroll
  for (int kt = 0; kt < 8; ++kt) {
    v16bf bb = bfrag_lds(buf, kt, lane);
#pragma unroll
    for (int mt = 0; mt < 8; ++mt)
      acc[mt] = wmma_bf16(afrag_glb(h0, 256, mt * 16, 0, kt, lane), bb, acc[mt]);
  }
  int n = nt * 16 + (lane & 15);
  float bias = bih[n] + bhh[n];
  int bhalf = (lane & 16) ? 8 : 0;
#pragma unroll
  for (int mt = 0; mt < 8; ++mt)
#pragma unroll
    for (int j = 0; j < 8; ++j)
      gates[(mt * 16 + j + bhalf) * 1024 + n] = acc[mt][j] + bias;
}

// ---------------- kernel 6: LSTM elementwise ----------------
__global__ void k_lstm(const float* __restrict__ gates, const float* __restrict__ c0,
                       float* __restrict__ outHN, float* __restrict__ outCN,
                       float* __restrict__ hnewWS) {
  int b = blockIdx.x, h = threadIdx.x;
  float gi = gates[b * 1024 + h];
  float gf = gates[b * 1024 + 256 + h];
  float gg = gates[b * 1024 + 512 + h];
  float go = gates[b * 1024 + 768 + h];
  float c  = c0[b * HID + h];
  float si = 1.f / (1.f + __expf(-gi));
  float sf = 1.f / (1.f + __expf(-gf));
  float so = 1.f / (1.f + __expf(-go));
  float cn = sf * c + si * tanhf(gg);
  float hn = so * tanhf(cn);
  outHN[b * HID + h] = hn;
  outCN[b * HID + h] = cn;
  hnewWS[b * HID + h] = hn;
}

// ---------------- kernel 6b: pack h_new into A-fragment layout ----------------
__global__ void k_packA(const float* __restrict__ hn, u32x8* __restrict__ ap) {
  int t    = blockIdx.x * blockDim.x + threadIdx.x;  // 2048 threads
  int lane = t & 31;
  int mt   = (t >> 5) & 7;
  int kt   = t >> 8;
  int m   = mt * 16 + (lane & 15);
  int khi = (lane & 16) ? 8 : 0;
  u32x8 d;
#pragma unroll
  for (int i = 0; i < 8; ++i) {
    int k = kt * 32 + (i & 3) * 2 + (i >> 2) * 16 + khi;
    f32x2 v = *reinterpret_cast<const f32x2*>(hn + (long)m * HID + k);
    d[i] = pack2(v[0], v[1]);
  }
  ap[t] = d;
}

// ---------------- kernel 7: predictions = h_new @ fc_w^T + fc_b ----------------
__global__ void __launch_bounds__(128) k_fc(const float* __restrict__ fcw,
                                            const float* __restrict__ fcb,
                                            const u32x8* __restrict__ ap,
                                            float* __restrict__ pred) {
  __shared__ unsigned short tile[4][16 * TSTRIDE];
  int lane = threadIdx.x & 31;
  int widx = threadIdx.x >> 5;
  int nt   = blockIdx.x * 4 + widx;
  if (nt >= 3142) return;                    // wave-uniform
  unsigned short* buf = tile[widx];
  // stage 16 fc_w rows (clamped at VOCAB) into LDS as bf16
#pragma unroll
  for (int it = 0; it < 32; ++it) {
    int idx = (it * 32 + lane) * 4;
    int row = idx >> 8;
    int col = idx & 255;
    int n = nt * 16 + row;
    if (n > VOCAB - 1) n = VOCAB - 1;
    f32x4 v = *reinterpret_cast<const f32x4*>(fcw + (long)n * HID + col);
    v4bf  r = __builtin_convertvector(v, v4bf);
    *reinterpret_cast<v4bf*>(&buf[row * TSTRIDE + col]) = r;
  }
  v8f acc[8];
  v8f zero = {};
#pragma unroll
  for (int q = 0; q < 8; ++q) acc[q] = zero;
#pragma unroll
  for (int kt = 0; kt < 8; ++kt) {
    v16bf bb = bfrag_lds(buf, kt, lane);
#pragma unroll
    for (int mt = 0; mt < 8; ++mt) {
      v16bf a = __builtin_bit_cast(v16bf, ap[(kt * 8 + mt) * 32 + lane]);
      acc[mt] = wmma_bf16(a, bb, acc[mt]);
    }
  }
  int n = nt * 16 + (lane & 15);
  bool ok = n < VOCAB;
  float bias = ok ? fcb[n] : 0.f;
  int bhalf = (lane & 16) ? 8 : 0;
#pragma unroll
  for (int mt = 0; mt < 8; ++mt)
#pragma unroll
    for (int j = 0; j < 8; ++j)
      if (ok)
        pred[(long)(mt * 16 + j + bhalf) * VOCAB + n] = acc[mt][j] + bias;
}

// ---------------- host launcher ----------------
extern "C" void kernel_launch(void* const* d_in, const int* in_sizes, int n_in,
                              void* d_out, int out_size, void* d_ws, size_t ws_size,
                              hipStream_t stream) {
  (void)in_sizes; (void)n_in; (void)out_size; (void)ws_size;
  const int*   x   = (const int*)d_in[0];
  const float* enc = (const float*)d_in[1];
  const float* h0  = (const float*)d_in[2];
  const float* c0  = (const float*)d_in[3];
  const float* emb = (const float*)d_in[4];
  const float* Wih = (const float*)d_in[5];
  const float* Whh = (const float*)d_in[6];
  const float* bih = (const float*)d_in[7];
  const float* bhh = (const float*)d_in[8];
  const float* fcw = (const float*)d_in[9];
  const float* fcb = (const float*)d_in[10];
  const float* W1  = (const float*)d_in[11];
  const float* W1b = (const float*)d_in[12];
  const float* W2  = (const float*)d_in[13];
  const float* W2b = (const float*)d_in[14];
  const float* Vw  = (const float*)d_in[15];
  const float* Vb  = (const float*)d_in[16];

  float* out = (float*)d_out;
  float* ws  = (float*)d_ws;

  float* uws   = ws + OFF_U;
  float* eBuf  = ws + OFF_E;
  float* ctxp  = ws + OFF_CTXP;
  float* rnn   = ws + OFF_RNN;
  float* gates = ws + OFF_GATES;
  float* hnew  = ws + OFF_HNEW;
  u32x8* w1p   = (u32x8*)(ws + OFF_W1P);
  u32x8* apack = (u32x8*)(ws + OFF_APACK);

  float* pred = out;                           // [128, 50257]
  float* oHN  = out + (size_t)BATCH * VOCAB;   // [1,128,256]
  float* oCN  = oHN + BATCH * HID;             // [1,128,256]
  float* attw = oCN + BATCH * HID;             // [128,2048,1]

  k_packW1     <<<16, 256, 0, stream>>>(W1, w1p);
  k_u          <<<32, 128, 0, stream>>>(h0, W2, W2b, W1b, uws);
  k_attn_score <<<4096, 128, 0, stream>>>(enc, uws, w1p, Vw, Vb, eBuf);
  k_softmax    <<<128, 256, 0, stream>>>(eBuf, attw);
  k_ctx_partial<<<1024, 256, 0, stream>>>(enc, attw, ctxp);
  k_rnn_in     <<<128, 256, 0, stream>>>(ctxp, emb, x, rnn);
  k_gates      <<<16, 128, 0, stream>>>(rnn, h0, Wih, Whh, bih, bhh, gates);
  k_lstm       <<<128, 256, 0, stream>>>(gates, c0, oHN, oCN, hnew);
  k_packA      <<<8, 256, 0, stream>>>(hnew, apack);
  k_fc         <<<786, 128, 0, stream>>>(fcw, fcb, apack, pred);
}
